// MultiHeadGraphAttention_42494406426624
// MI455X (gfx1250) — compile-verified, matrix operand order checked by reading, not run on hardware
//
#include <hip/hip_runtime.h>
#include <math.h>

// ---------------------------------------------------------------------------
// Multi-head graph attention, CDNA5 (gfx1250, wave32, WMMA).
// Dense GEMMs: v_wmma_f32_16x16x32_f16, f16 operands / f32 accumulate.
//   - Activations staged in LDS, padded stride (HID+8) to break the 256B
//     bank period; fragment loads are 2 x b128 per lane (ISA 7.12.2 layout:
//     two contiguous 8xf16 runs at K = 8*half+{0..7} and +16).
//   - Weights packed once per launch to transposed f16 in workspace global
//     memory (Wt[n*K+k]); B fragments load straight from L2 with the same
//     2 x global_load_b128 pattern (220KB total, resident in 192MB L2).
// Irregular segment-softmax / scatter: u32-encoded atomicMax + f32 atomicAdd.
// ---------------------------------------------------------------------------

typedef __attribute__((ext_vector_type(16))) _Float16 v16h;
typedef __attribute__((ext_vector_type(8)))  _Float16 v8h;
typedef __attribute__((ext_vector_type(8)))  float    v8f;

#define NN      50000
#define NE      800000
#define NODE_IN 64
#define EDGE_IN 32
#define HID     128
#define NHEADS  4
#define HD      32
#define RSQRT_HD 0.1767766953f   // 1/sqrt(32)

// padded LDS strides (must stay multiples of 8 f16 for 16B-aligned b128 rows)
#define LDA_T (HID + 8)      // 136 : 128-wide activation tiles
#define LDA_N (NODE_IN + 8)  // 72  : node input tile
#define LDA_E (EDGE_IN + 8)  // 40  : edge input tile

__device__ __forceinline__ int lane_id() { return threadIdx.x & 31; }
__device__ __forceinline__ int wave_id() { return threadIdx.x >> 5; }
__device__ __forceinline__ v8f vzero8() { v8f z = {}; return z; }

// --- unified 16x32 f16 fragment load (A from LDS rows, B from global Wt) ----
// lane L: row = L%16, half = L/16; data = two contiguous 8xf16 (16B) runs.
__device__ __forceinline__ v16h load_frag(const _Float16* p0, int ld) {
  const int lane = lane_id();
  const int row  = lane & 15;
  const int half = lane >> 4;
  const _Float16* p = p0 + row * ld + half * 8;
  union { v16h v; v8h h[2]; } u;
  u.h[0] = *(const v8h*)(p);        // K = 8*half + 0..7
  u.h[1] = *(const v8h*)(p + 16);   // K = 8*half + 16..23
  return u.v;
}

// One wave computes a 16 x 128 output stripe: 8 WMMA tiles, K-loop in 32s.
// A: LDS activation tile (stride lda). Wt: transposed f16 weights in global,
// row n has contiguous K (stride = K).
__device__ __forceinline__ void gemm_wave_16x128(const _Float16* A, int lda, int K,
                                                 const _Float16* __restrict__ Wt,
                                                 v8f acc[8]) {
  for (int kt = 0; kt < K; kt += 32) {
    v16h a = load_frag(A + kt, lda);
#pragma unroll
    for (int j = 0; j < 8; ++j) {
      v16h b = load_frag(Wt + (j * 16) * K + kt, K);
      acc[j] = __builtin_amdgcn_wmma_f32_16x16x32_f16(
          /*neg_a=*/false, a, /*neg_b=*/false, b,
          /*c_mod=*/(short)0, acc[j], /*reuse_a=*/false, /*reuse_b=*/false);
    }
  }
}

// Monotone uint encoding of float for atomicMax-based segment max.
__device__ __forceinline__ unsigned enc_f(float f) {
  unsigned u = __float_as_uint(f);
  return (u & 0x80000000u) ? ~u : (u | 0x80000000u);
}
__device__ __forceinline__ float dec_f(unsigned u) {
  u = (u & 0x80000000u) ? (u & 0x7fffffffu) : ~u;
  return __uint_as_float(u);
}

// ---------------------------------------------------------------------------
// Kernel 0: pack one weight matrix [K, N] f32 -> transposed f16 Wt[n*K + k].
// ---------------------------------------------------------------------------
__global__ void ga_pack_w(const float* __restrict__ src, _Float16* __restrict__ dst,
                          int K, int Ncols) {
  const int i = blockIdx.x * blockDim.x + threadIdx.x;
  if (i >= K * Ncols) return;
  const int k = i / Ncols;
  const int n = i % Ncols;
  dst[n * K + k] = (_Float16)src[i];
}

// ---------------------------------------------------------------------------
// Kernel 1: fused node encoder + Q/K/V.  64 rows / block, 4 waves (128 thr).
// LDS: sA (input tile, then h tile) | sT (relu intermediate)  ~34KB total.
// ---------------------------------------------------------------------------
__global__ __launch_bounds__(128)
void ga_node_encode_qkv(const float* __restrict__ x,
                        const _Float16* __restrict__ w1t, const float* b1,
                        const _Float16* __restrict__ w2t, const float* b2,
                        const _Float16* __restrict__ wqt, const float* bq,
                        const _Float16* __restrict__ wkt, const float* bk,
                        const _Float16* __restrict__ wvt, const float* bv,
                        float* __restrict__ gH, float* __restrict__ gQ,
                        float* __restrict__ gK, float* __restrict__ gV) {
  __shared__ _Float16 sA[64 * LDA_T];  // input tile (64 x LDA_N region), later h
  __shared__ _Float16 sT[64 * LDA_T];  // relu intermediate

  const int tid   = threadIdx.x;
  const int row0  = blockIdx.x * 64;
  const int w     = wave_id();
  const int lane  = lane_id();
  const int cn    = lane & 15;
  const int rhalf = lane >> 4;

  // stage input tile (f32 -> f16), zero-pad out-of-range rows
  for (int i = tid; i < 64 * NODE_IN; i += 128) {
    const int r = i / NODE_IN, c = i % NODE_IN;
    sA[r * LDA_N + c] =
        (row0 + r < NN) ? (_Float16)x[(size_t)(row0 + r) * NODE_IN + c]
                        : (_Float16)0.0f;
  }
  __syncthreads();

  v8f acc[8];

  // ---- GEMM1: t = relu(x @ W1 + b1) ----
#pragma unroll
  for (int j = 0; j < 8; ++j) acc[j] = vzero8();
  gemm_wave_16x128(sA + (w * 16) * LDA_N, LDA_N, NODE_IN, w1t, acc);
#pragma unroll
  for (int j = 0; j < 8; ++j) {
    const int col = j * 16 + cn;
    const float bias = b1[col];
#pragma unroll
    for (int r = 0; r < 8; ++r) {
      const int m = r + 8 * rhalf;
      sT[(w * 16 + m) * LDA_T + col] = (_Float16)fmaxf(acc[j][r] + bias, 0.0f);
    }
  }
  __syncthreads();

  // ---- GEMM2: h = t @ W2 + b2  (sA reused as h tile; input region dead) ----
#pragma unroll
  for (int j = 0; j < 8; ++j) acc[j] = vzero8();
  gemm_wave_16x128(sT + (w * 16) * LDA_T, LDA_T, HID, w2t, acc);
#pragma unroll
  for (int j = 0; j < 8; ++j) {
    const int col = j * 16 + cn;
    const float bias = b2[col];
#pragma unroll
    for (int r = 0; r < 8; ++r) {
      const int m = r + 8 * rhalf;
      const float v = acc[j][r] + bias;
      sA[(w * 16 + m) * LDA_T + col] = (_Float16)v;
      const int row = row0 + w * 16 + m;
      if (row < NN) gH[(size_t)row * HID + col] = v;
    }
  }
  __syncthreads();

  // ---- Q / K / V projections from the h tile (weights direct from L2) ----
  const _Float16* Ws[3] = {wqt, wkt, wvt};
  const float*    Bs[3] = {bq, bk, bv};
  float*          Os[3] = {gQ, gK, gV};
#pragma unroll
  for (int p = 0; p < 3; ++p) {
#pragma unroll
    for (int j = 0; j < 8; ++j) acc[j] = vzero8();
    gemm_wave_16x128(sA + (w * 16) * LDA_T, LDA_T, HID, Ws[p], acc);
#pragma unroll
    for (int j = 0; j < 8; ++j) {
      const int col = j * 16 + cn;
      const float bias = Bs[p][col];
#pragma unroll
      for (int r = 0; r < 8; ++r) {
        const int row = row0 + w * 16 + r + 8 * rhalf;
        if (row < NN) Os[p][(size_t)row * HID + col] = acc[j][r] + bias;
      }
    }
  }
}

// ---------------------------------------------------------------------------
// Kernel 2: edge encoder + per-head edge bias eb = e @ ea_w + ea_b.
// ---------------------------------------------------------------------------
__global__ __launch_bounds__(128)
void ga_edge_encode(const float* __restrict__ ef,
                    const _Float16* __restrict__ w1t, const float* b1,
                    const _Float16* __restrict__ w2t, const float* b2,
                    const float* eaw, const float* eab,
                    float* __restrict__ gEB) {
  __shared__ _Float16 sA[64 * LDA_T];  // input tile, later e tile
  __shared__ _Float16 sT[64 * LDA_T];  // relu intermediate (sEA aliased later)

  const int tid   = threadIdx.x;
  const int row0  = blockIdx.x * 64;
  const int w     = wave_id();
  const int lane  = lane_id();
  const int cn    = lane & 15;
  const int rhalf = lane >> 4;

  for (int i = tid; i < 64 * EDGE_IN; i += 128) {
    const int r = i / EDGE_IN, c = i % EDGE_IN;
    sA[r * LDA_E + c] =
        (row0 + r < NE) ? (_Float16)ef[(size_t)(row0 + r) * EDGE_IN + c]
                        : (_Float16)0.0f;
  }
  __syncthreads();

  v8f acc[8];

  // GEMM1: t = relu(ef @ W1 + b1), K = 32
#pragma unroll
  for (int j = 0; j < 8; ++j) acc[j] = vzero8();
  gemm_wave_16x128(sA + (w * 16) * LDA_E, LDA_E, EDGE_IN, w1t, acc);
#pragma unroll
  for (int j = 0; j < 8; ++j) {
    const int col = j * 16 + cn;
    const float bias = b1[col];
#pragma unroll
    for (int r = 0; r < 8; ++r) {
      const int m = r + 8 * rhalf;
      sT[(w * 16 + m) * LDA_T + col] = (_Float16)fmaxf(acc[j][r] + bias, 0.0f);
    }
  }
  __syncthreads();

  // GEMM2: e = t @ W2 + b2  (into sA; input tile dead)
#pragma unroll
  for (int j = 0; j < 8; ++j) acc[j] = vzero8();
  gemm_wave_16x128(sT + (w * 16) * LDA_T, LDA_T, HID, w2t, acc);
#pragma unroll
  for (int j = 0; j < 8; ++j) {
    const int col = j * 16 + cn;
    const float bias = b2[col];
#pragma unroll
    for (int r = 0; r < 8; ++r) {
      const int m = r + 8 * rhalf;
      sA[(w * 16 + m) * LDA_T + col] = (_Float16)(acc[j][r] + bias);
    }
  }
  __syncthreads();

  // stage ea_w (128x4 f32) into sT's space (sT dead now)
  float* sEA = (float*)sT;
  for (int i = tid; i < HID * NHEADS; i += 128) sEA[i] = eaw[i];
  __syncthreads();

  // eb[row, h] = e[row,:] . ea_w[:,h] + ea_b[h]; thread -> (row, head-pair)
  {
    const int r  = tid & 63;
    const int h0 = (tid >> 6) * 2;
    const int e  = row0 + r;
    if (e < NE) {
#pragma unroll
      for (int hh = 0; hh < 2; ++hh) {
        const int h = h0 + hh;
        float s = eab[h];
        for (int k = 0; k < HID; ++k)
          s += (float)sA[r * LDA_T + k] * sEA[k * NHEADS + h];
        gEB[(size_t)e * NHEADS + h] = s;
      }
    }
  }
}

// ---------------------------------------------------------------------------
// Kernel 3: init segment-max (enc(-inf)), denom, messages.
// ---------------------------------------------------------------------------
__global__ void ga_init(unsigned* __restrict__ gMax, float* __restrict__ gDenom,
                        float* __restrict__ gMsg) {
  const int i = blockIdx.x * blockDim.x + threadIdx.x;
  if (i < NN * NHEADS) { gMax[i] = 0x007FFFFFu; gDenom[i] = 0.0f; }  // enc(-inf)
  if (i < NN * HID)    gMsg[i] = 0.0f;
}

// ---------------------------------------------------------------------------
// Kernel 4: per-(edge,head) score = dot(Q[src],K[dst])/sqrt(HD) + eb, and
// segment max via monotone-uint atomicMax on src.
// ---------------------------------------------------------------------------
__global__ void ga_scores(const int* __restrict__ eidx,
                          const float* __restrict__ gQ, const float* __restrict__ gK,
                          float* __restrict__ gScore,    // in: eb, out: score
                          unsigned* __restrict__ gMax) {
  const int idx = blockIdx.x * blockDim.x + threadIdx.x;
  if (idx >= NE * NHEADS) return;
  const int e = idx >> 2, h = idx & 3;
  const int src = eidx[e];
  const int dst = eidx[NE + e];
  const float4* q = (const float4*)(gQ + (size_t)src * HID + h * HD);
  const float4* k = (const float4*)(gK + (size_t)dst * HID + h * HD);
  float s = 0.0f;
#pragma unroll
  for (int d = 0; d < HD / 4; ++d) {
    const float4 a = q[d], b = k[d];
    s += a.x * b.x + a.y * b.y + a.z * b.z + a.w * b.w;
  }
  s = s * RSQRT_HD + gScore[idx];        // + edge contribution (prior mask == 1)
  gScore[idx] = s;
  atomicMax(gMax + (size_t)src * NHEADS + h, enc_f(s));
}

// ---------------------------------------------------------------------------
// Kernel 5: ex = exp(score - segmax[src]); denom[src] += ex (atomic).
// ---------------------------------------------------------------------------
__global__ void ga_expsum(const int* __restrict__ eidx,
                          const unsigned* __restrict__ gMax,
                          float* __restrict__ gScore, float* __restrict__ gDenom) {
  const int idx = blockIdx.x * blockDim.x + threadIdx.x;
  if (idx >= NE * NHEADS) return;
  const int e = idx >> 2, h = idx & 3;
  const int src = eidx[e];
  const float m  = dec_f(gMax[(size_t)src * NHEADS + h]);  // finite: src has edges
  const float ex = expf(gScore[idx] - m);
  gScore[idx] = ex;
  atomicAdd(gDenom + (size_t)src * NHEADS + h, ex);
}

// ---------------------------------------------------------------------------
// Kernel 6: attn = ex/denom[src]; messages[dst] += V[src]*attn (atomic scatter).
// Also writes attn_weights into the second output region.
// ---------------------------------------------------------------------------
__global__ void ga_scatter(const int* __restrict__ eidx,
                           const float* __restrict__ gDenom,
                           const float* __restrict__ gScore,
                           const float* __restrict__ gV,
                           float* __restrict__ gMsg, float* __restrict__ gAttn) {
  const int idx = blockIdx.x * blockDim.x + threadIdx.x;
  if (idx >= NE * NHEADS) return;
  const int e = idx >> 2, h = idx & 3;
  const int src = eidx[e];
  const int dst = eidx[NE + e];
  const float wgt = gScore[idx] / gDenom[(size_t)src * NHEADS + h];
  gAttn[idx] = wgt;
  const float* v = gV + (size_t)src * HID + h * HD;
  float* m = gMsg + (size_t)dst * HID + h * HD;
#pragma unroll
  for (int d = 0; d < HD; ++d) atomicAdd(m + d, v[d] * wgt);
}

// ---------------------------------------------------------------------------
// Kernel 7: output = h + messages @ op_w + op_b   (WMMA GEMM + residual)
// ---------------------------------------------------------------------------
__global__ __launch_bounds__(128)
void ga_output(const float* __restrict__ gMsg,
               const _Float16* __restrict__ opwt, const float* opb,
               const float* __restrict__ gH, float* __restrict__ gOut) {
  __shared__ _Float16 sM[64 * LDA_T];

  const int tid   = threadIdx.x;
  const int row0  = blockIdx.x * 64;
  const int w     = wave_id();
  const int lane  = lane_id();
  const int cn    = lane & 15;
  const int rhalf = lane >> 4;

  for (int i = tid; i < 64 * HID; i += 128) {
    const int r = i / HID, c = i % HID;
    sM[r * LDA_T + c] =
        (row0 + r < NN) ? (_Float16)gMsg[(size_t)(row0 + r) * HID + c]
                        : (_Float16)0.0f;
  }
  __syncthreads();

  v8f acc[8];
#pragma unroll
  for (int j = 0; j < 8; ++j) acc[j] = vzero8();
  gemm_wave_16x128(sM + (w * 16) * LDA_T, LDA_T, HID, opwt, acc);
#pragma unroll
  for (int j = 0; j < 8; ++j) {
    const int col = j * 16 + cn;
    const float bias = opb[col];
#pragma unroll
    for (int r = 0; r < 8; ++r) {
      const int row = row0 + w * 16 + r + 8 * rhalf;
      if (row < NN)
        gOut[(size_t)row * HID + col] =
            gH[(size_t)row * HID + col] + acc[j][r] + bias;
    }
  }
}

// ---------------------------------------------------------------------------
extern "C" void kernel_launch(void* const* d_in, const int* in_sizes, int n_in,
                              void* d_out, int out_size, void* d_ws, size_t ws_size,
                              hipStream_t stream) {
  const float* node_feat = (const float*)d_in[0];
  const int*   eidx      = (const int*)d_in[1];   // [2, E] int32 (JAX x64 off)
  const float* edge_feat = (const float*)d_in[2];
  // d_in[3] = ball_progress (unused by the reference math)
  const float* ne_w1 = (const float*)d_in[4];
  const float* ne_b1 = (const float*)d_in[5];
  const float* ne_w2 = (const float*)d_in[6];
  const float* ne_b2 = (const float*)d_in[7];
  const float* ee_w1 = (const float*)d_in[8];
  const float* ee_b1 = (const float*)d_in[9];
  const float* ee_w2 = (const float*)d_in[10];
  const float* ee_b2 = (const float*)d_in[11];
  const float* wq    = (const float*)d_in[12];
  const float* bq    = (const float*)d_in[13];
  const float* wk    = (const float*)d_in[14];
  const float* bk    = (const float*)d_in[15];
  const float* wv    = (const float*)d_in[16];
  const float* bv    = (const float*)d_in[17];
  const float* ea_w  = (const float*)d_in[18];
  const float* ea_b  = (const float*)d_in[19];
  const float* op_w  = (const float*)d_in[20];
  const float* op_b  = (const float*)d_in[21];

  float* out     = (float*)d_out;                 // [N, HID]
  float* outAttn = out + (size_t)NN * HID;        // [E, HEADS]

  // workspace carve-out (256B-aligned slices)
  char* ws = (char*)d_ws;
  size_t off = 0;
  auto carve = [&](size_t bytes) -> void* {
    void* p = ws + off;
    off += (bytes + 255) & ~(size_t)255;
    return p;
  };
  // packed transposed f16 weights (resident in L2 across all GEMM blocks)
  _Float16* w1t  = (_Float16*)carve((size_t)NODE_IN * HID * 2);
  _Float16* w2t  = (_Float16*)carve((size_t)HID * HID * 2);
  _Float16* wqt  = (_Float16*)carve((size_t)HID * HID * 2);
  _Float16* wkt  = (_Float16*)carve((size_t)HID * HID * 2);
  _Float16* wvt  = (_Float16*)carve((size_t)HID * HID * 2);
  _Float16* e1t  = (_Float16*)carve((size_t)EDGE_IN * HID * 2);
  _Float16* e2t  = (_Float16*)carve((size_t)HID * HID * 2);
  _Float16* opt  = (_Float16*)carve((size_t)HID * HID * 2);
  // activations / softmax state
  float*    gH     = (float*)carve((size_t)NN * HID * 4);
  float*    gQ     = (float*)carve((size_t)NN * HID * 4);
  float*    gK     = (float*)carve((size_t)NN * HID * 4);
  float*    gV     = (float*)carve((size_t)NN * HID * 4);
  float*    gEB    = (float*)carve((size_t)NE * NHEADS * 4); // eb -> score -> ex
  unsigned* gMax   = (unsigned*)carve((size_t)NN * NHEADS * 4);
  float*    gDenom = (float*)carve((size_t)NN * NHEADS * 4);
  float*    gMsg   = (float*)carve((size_t)NN * HID * 4);

  auto pack = [&](const float* src, _Float16* dst, int K, int Ncols) {
    const int tot = K * Ncols;
    ga_pack_w<<<(tot + 255) / 256, 256, 0, stream>>>(src, dst, K, Ncols);
  };
  pack(ne_w1, w1t, NODE_IN, HID);
  pack(ne_w2, w2t, HID, HID);
  pack(wq,    wqt, HID, HID);
  pack(wk,    wkt, HID, HID);
  pack(wv,    wvt, HID, HID);
  pack(ee_w1, e1t, EDGE_IN, HID);
  pack(ee_w2, e2t, HID, HID);
  pack(op_w,  opt, HID, HID);

  const int nodeBlocks = (NN + 63) / 64;
  const int edgeBlocks = (NE + 63) / 64;
  const int ehThreads  = NE * NHEADS;
  const int initSpan   = NN * HID;

  ga_node_encode_qkv<<<nodeBlocks, 128, 0, stream>>>(
      node_feat, w1t, ne_b1, w2t, ne_b2, wqt, bq, wkt, bk, wvt, bv,
      gH, gQ, gK, gV);
  ga_edge_encode<<<edgeBlocks, 128, 0, stream>>>(
      edge_feat, e1t, ee_b1, e2t, ee_b2, ea_w, ea_b, gEB);
  ga_init<<<(initSpan + 255) / 256, 256, 0, stream>>>(gMax, gDenom, gMsg);
  ga_scores<<<(ehThreads + 255) / 256, 256, 0, stream>>>(eidx, gQ, gK, gEB, gMax);
  ga_expsum<<<(ehThreads + 255) / 256, 256, 0, stream>>>(eidx, gMax, gEB, gDenom);
  ga_scatter<<<(ehThreads + 255) / 256, 256, 0, stream>>>(eidx, gDenom, gEB, gV,
                                                          gMsg, outAttn);
  ga_output<<<nodeBlocks, 128, 0, stream>>>(gMsg, opt, op_b, gH, out);
}
